// InputLinearizationRnn2_59141699666502
// MI455X (gfx1250) — matrix operand order repaired and checked by reading, not compiled, not run on hardware
//
#include <hip/hip_runtime.h>
#include <hip/hip_bf16.h>
#include <math.h>

// Problem constants (match the JAX reference)
#define BATCH 256
#define NSTEP 1024
#define NSD   128   // NS state dim
#define NDD   32    // ND input dim
#define NED   32    // NE output dim
#define NWD   64    // NW nonlinearity dim

typedef __attribute__((ext_vector_type(16))) __bf16 v16bf;
typedef __attribute__((ext_vector_type(8)))  float  v8f;

// Padded row strides in bf16 elements (multiples of 8 -> 16B aligned rows,
// and 272/144/80-byte strides break the 64-bank LDS conflict pattern)
#define PK128 136
#define PK64  72
#define PK32  40

// LDS layout (offsets in bf16 elements; every offset is a multiple of 8)
#define OFF_A    0                              // 128 x PK128
#define OFF_C2   17408                          //  64 x PK128
#define OFF_C1   26112                          //  32 x PK128
#define OFF_B2   30464                          // 128 x PK64
#define OFF_D12  39680                          //  32 x PK64
#define OFF_B1   41984                          // 128 x PK32
#define OFF_D21  47104                          //  64 x PK32
#define OFF_D11  49664                          //  32 x PK32
#define OFF_X    50944                          //  16 x PK128 (state tile)
#define OFF_W    53120                          //  16 x PK64  (w tile)
#define SMEM_ELEMS 54272
#define SMEM_BYTES (SMEM_ELEMS * 2)             // 108,544 B < 320 KB/WGP

union AccU  { v8f v; float f[8]; };
union FragU { v16bf v; uint4 q[2]; __bf16 e[16]; };

#define WMMA_BF16(Af, Bf, Cv) \
  __builtin_amdgcn_wmma_f32_16x16x32_bf16(false, (Af), false, (Bf), (short)0, (Cv), false, false)

// tanh(x) = 1 - 2/(exp2(2x*log2e)+1); exact at +/-inf, monotone, ~ulp-level
// error that is far below the bf16 quantization applied right after.
// Lowered to v_exp_f32 + v_rcp_f32 (TRANS) + 2 VALU, vs ~20 ops for libm tanhf.
__device__ __forceinline__ float fast_tanh(float x) {
  float e = __builtin_amdgcn_exp2f(x * 2.885390081777927f); // 2*log2(e)
  float r = __builtin_amdgcn_rcpf(e + 1.0f);
  return 1.0f - 2.0f * r;
}

// A-fragment (16xK tile as WMMA A operand), source row-major bf16 in LDS.
// ISA 16-bit A layout: lane L (m=L&15, hl=L>>4); halves 0-7 = K[kBase+8hl .. +7],
// halves 8-15 = K[kBase+16+8hl .. +7]  -> two contiguous 16B chunks per lane.
__device__ __forceinline__ v16bf load_a_frag(const __bf16* base, int stride,
                                             int m, int hl, int kBase) {
  const __bf16* p = base + m * stride + kBase + 8 * hl;
  FragU u;
  u.q[0] = *reinterpret_cast<const uint4*>(p);
  u.q[1] = *reinterpret_cast<const uint4*>(p + 16);
  return u.v;
}

// B-fragment (Kx16 tile of B = W^T as WMMA B operand). W stored row-major
// [out][in] in LDS; column n of B is row (nTile*16+n) of W, contiguous in K.
// Lane half hl holds K=kb*32+8hl..+7 (halves 0-7) and K=kb*32+16+8hl..+7 (halves 8-15).
__device__ __forceinline__ v16bf load_b_frag(const __bf16* wbase, int stride,
                                             int n, int hl, int nTile, int kb) {
  const __bf16* p = wbase + (nTile * 16 + n) * stride + kb * 32 + 8 * hl;
  FragU u;
  u.q[0] = *reinterpret_cast<const uint4*>(p);
  u.q[1] = *reinterpret_cast<const uint4*>(p + 16);
  return u.v;
}

__device__ __forceinline__ void stage_weight(const float* g, __bf16* s,
                                             int rows, int cols, int stride, int tid) {
  for (int i = tid; i < rows * cols; i += 32) {
    int r = i / cols, c = i - r * cols;
    s[r * stride + c] = (__bf16)g[i];
  }
}

extern "C" __global__ __launch_bounds__(32, 1)
void lure_scan_wmma_kernel(const float* __restrict__ x0, const float* __restrict__ us,
                           const float* __restrict__ A,  const float* __restrict__ B1,
                           const float* __restrict__ B2, const float* __restrict__ C1,
                           const float* __restrict__ D11,const float* __restrict__ D12,
                           const float* __restrict__ C2, const float* __restrict__ D21,
                           float* __restrict__ out) {
  extern __shared__ __align__(16) __bf16 smem[];
  const int tid = threadIdx.x;
  const int m   = tid & 15;   // A-frag row / B-frag column / D-matrix column
  const int hl  = tid >> 4;   // half-wave select
  const int b0  = blockIdx.x * 16;

  __bf16* sA   = smem + OFF_A;
  __bf16* sC2  = smem + OFF_C2;
  __bf16* sC1  = smem + OFF_C1;
  __bf16* sB2  = smem + OFF_B2;
  __bf16* sD12 = smem + OFF_D12;
  __bf16* sB1  = smem + OFF_B1;
  __bf16* sD21 = smem + OFF_D21;
  __bf16* sD11 = smem + OFF_D11;
  __bf16* sx   = smem + OFF_X;
  __bf16* sw   = smem + OFF_W;

  // One-time: stage all weights (fp32 -> bf16) into LDS, row-major [out][in].
  stage_weight(A,   sA,   128, 128, PK128, tid);
  stage_weight(C2,  sC2,   64, 128, PK128, tid);
  stage_weight(C1,  sC1,   32, 128, PK128, tid);
  stage_weight(B2,  sB2,  128,  64, PK64,  tid);
  stage_weight(D12, sD12,  32,  64, PK64,  tid);
  stage_weight(B1,  sB1,  128,  32, PK32,  tid);
  stage_weight(D21, sD21,  64,  32, PK32,  tid);
  stage_weight(D11, sD11,  32,  32, PK32,  tid);

  // Initial state tile: x0[b0+r, :, 0] -> sx (bf16)
  for (int i = tid; i < 16 * NSD; i += 32) {
    int r = i >> 7, c = i & 127;
    sx[r * PK128 + c] = (__bf16)x0[(size_t)(b0 + r) * NSD + c];
  }
  __syncthreads();

  // State x held as 4 A-fragments (16 x 128 bf16) in VGPRs.
  v16bf xf[4];
#pragma unroll
  for (int kb = 0; kb < 4; ++kb) xf[kb] = load_a_frag(sx, PK128, m, hl, kb * 32);

#pragma unroll 1
  for (int k = 0; k < NSTEP; ++k) {
    // ---- u_k tile as A-fragment (fp32 global -> bf16 regs), K = 32
    const float* up = us + ((size_t)(b0 + m) * NSTEP + k) * NDD;
    FragU uf;
#pragma unroll
    for (int i = 0; i < 8; ++i) {
      uf.e[i]     = (__bf16)up[8 * hl + i];
      uf.e[8 + i] = (__bf16)up[16 + 8 * hl + i];
    }

    // ---- Phase 1: w = tanh(x C2^T + u D21^T)  (4 n-tiles of 16, K=128+32)
#pragma unroll
    for (int nt = 0; nt < 4; ++nt) {
      // Batch-preload B fragments so the ds_loads overlap the WMMA chain.
      v16bf bf[4];
#pragma unroll
      for (int kb = 0; kb < 4; ++kb) bf[kb] = load_b_frag(sC2, PK128, m, hl, nt, kb);
      v16bf bu = load_b_frag(sD21, PK32, m, hl, nt, 0);
      AccU acc; acc.v = {};
#pragma unroll
      for (int kb = 0; kb < 4; ++kb) acc.v = WMMA_BF16(xf[kb], bf[kb], acc.v);
      acc.v = WMMA_BF16(uf.v, bu, acc.v);
      // D-layout: vgpr j -> row (j + 8*hl), lane -> col m. Store bf16 w to LDS.
#pragma unroll
      for (int j = 0; j < 8; ++j)
        sw[(j + 8 * hl) * PK64 + nt * 16 + m] = (__bf16)fast_tanh(acc.f[j]);
    }
    __syncthreads();

    v16bf wfr[2];
#pragma unroll
    for (int kw = 0; kw < 2; ++kw) wfr[kw] = load_a_frag(sw, PK64, m, hl, kw * 32);

    // ---- Phase 2: x_new = x A^T + u B1^T + w B2^T  (8 n-tiles, K=128+32+64)
#pragma unroll
    for (int nt = 0; nt < 8; ++nt) {
      v16bf bf[4];
#pragma unroll
      for (int kb = 0; kb < 4; ++kb) bf[kb] = load_b_frag(sA, PK128, m, hl, nt, kb);
      AccU acc; acc.v = {};
#pragma unroll
      for (int kb = 0; kb < 4; ++kb) acc.v = WMMA_BF16(xf[kb], bf[kb], acc.v);
      v16bf b1f = load_b_frag(sB1, PK32, m, hl, nt, 0);
      v16bf b2f0 = load_b_frag(sB2, PK64, m, hl, nt, 0);
      v16bf b2f1 = load_b_frag(sB2, PK64, m, hl, nt, 1);
      acc.v = WMMA_BF16(uf.v, b1f, acc.v);
      acc.v = WMMA_BF16(wfr[0], b2f0, acc.v);
      acc.v = WMMA_BF16(wfr[1], b2f1, acc.v);
      // x is fully held in xf registers; overwrite sx in place with x_new.
#pragma unroll
      for (int j = 0; j < 8; ++j)
        sx[(j + 8 * hl) * PK128 + nt * 16 + m] = (__bf16)acc.f[j];
    }
    __syncthreads();

    // Reload x_new as A-fragments (becomes next step's x).
#pragma unroll
    for (int kb = 0; kb < 4; ++kb) xf[kb] = load_a_frag(sx, PK128, m, hl, kb * 32);

    // ---- Phase 3: y = x_new C1^T + u D11^T + w D12^T  (2 n-tiles)
#pragma unroll
    for (int nt = 0; nt < 2; ++nt) {
      v16bf bf[4];
#pragma unroll
      for (int kb = 0; kb < 4; ++kb) bf[kb] = load_b_frag(sC1, PK128, m, hl, nt, kb);
      AccU acc; acc.v = {};
#pragma unroll
      for (int kb = 0; kb < 4; ++kb) acc.v = WMMA_BF16(xf[kb], bf[kb], acc.v);
      v16bf bd11 = load_b_frag(sD11, PK32, m, hl, nt, 0);
      v16bf bd12a = load_b_frag(sD12, PK64, m, hl, nt, 0);
      v16bf bd12b = load_b_frag(sD12, PK64, m, hl, nt, 1);
      acc.v = WMMA_BF16(uf.v, bd11, acc.v);
      acc.v = WMMA_BF16(wfr[0], bd12a, acc.v);
      acc.v = WMMA_BF16(wfr[1], bd12b, acc.v);
      // out[b, k, e, 0] with b = b0 + j + 8*hl, e = nt*16 + m
#pragma unroll
      for (int j = 0; j < 8; ++j)
        out[((size_t)(b0 + j + 8 * hl) * NSTEP + k) * NED + nt * 16 + m] = acc.f[j];
    }
  }
}

extern "C" void kernel_launch(void* const* d_in, const int* in_sizes, int n_in,
                              void* d_out, int out_size, void* d_ws, size_t ws_size,
                              hipStream_t stream) {
  (void)in_sizes; (void)n_in; (void)d_ws; (void)ws_size; (void)out_size;
  const float* x0  = (const float*)d_in[0];
  const float* us  = (const float*)d_in[1];
  const float* A   = (const float*)d_in[2];
  const float* B1  = (const float*)d_in[3];
  const float* B2  = (const float*)d_in[4];
  const float* C1  = (const float*)d_in[5];
  const float* D11 = (const float*)d_in[6];
  const float* D12 = (const float*)d_in[7];
  const float* C2  = (const float*)d_in[8];
  const float* D21 = (const float*)d_in[9];
  float* out = (float*)d_out;

  // Allow >64KB dynamic LDS (idempotent; not a stream op, safe under capture).
  (void)hipFuncSetAttribute((const void*)lure_scan_wmma_kernel,
                            hipFuncAttributeMaxDynamicSharedMemorySize, SMEM_BYTES);

  // 16 batch tiles of M=16, one wave32 per tile, one workgroup per wave.
  lure_scan_wmma_kernel<<<BATCH / 16, 32, SMEM_BYTES, stream>>>(
      x0, us, A, B1, B2, C1, D11, D12, C2, D21, out);
}
